// SemanticEmbeddingBranch_58025008169448
// MI455X (gfx1250) — matrix operand order repaired and checked by reading, not compiled, not run on hardware
//
#include <hip/hip_runtime.h>

// ---------------------------------------------------------------------------
// Dynamic depthwise 7x7 conv (softmax kernels) + LayerNorm(C) + ReLU + residual
// MI455X / gfx1250: conv runs on v_wmma_f32_16x16x32_bf16 via a Toeplitz
// banded-matmul formulation; LN is fused using the 320KB WGP LDS.
// ---------------------------------------------------------------------------

typedef __bf16 bf16_t;
typedef __attribute__((ext_vector_type(16))) __bf16 v16bf;
typedef __attribute__((ext_vector_type(8)))  __bf16 v8bf;
typedef __attribute__((ext_vector_type(8)))  float  v8f;

#define B_   16
#define C_   256
#define H_   96
#define W_   96
#define HP   102               // 96 + 3 top + 3 bottom halo rows
#define WP   104               // padded row stride in bf16 elems (16B aligned rows)
#define PATCH_ELEMS (HP * WP)  // 10608 bf16 per (b,c)
#define ZPATCH_ELEMS (PATCH_ELEMS + 64) // zero patch + slack for chunk-2 overhang
#define BPKT_HALVES (7 * 32 * 16) // 3584 bf16 per (b,c): 7 Toeplitz B-matrices
#define NWAVES 16
#define CH_PER_WAVE (C_ / NWAVES) // 16

// ---------------------------------------------------------------------------
// Prep 1: convert x (f32) into zero-padded bf16 image with a 3-pixel halo so
// the WMMA A-operand loads in the hot kernel are unconditional and 16B aligned.
// Block index B_*C_ writes the shared all-zero "phantom channel" patch that the
// upper half-wave reads for its (mathematically dead) second A chunk.
// ---------------------------------------------------------------------------
__global__ void pad_convert_kernel(const float* __restrict__ x,
                                   bf16_t* __restrict__ xpad,
                                   bf16_t* __restrict__ zpatch) {
  const int bc = blockIdx.x;                      // b*C + c, or B_*C_ for zeros
  if (bc == B_ * C_) {
    for (int idx = threadIdx.x; idx < ZPATCH_ELEMS; idx += blockDim.x)
      zpatch[idx] = (bf16_t)0.0f;
    return;
  }
  const float* src = x + (size_t)bc * (H_ * W_);
  bf16_t* dst = xpad + (size_t)bc * PATCH_ELEMS;
  for (int idx = threadIdx.x; idx < PATCH_ELEMS; idx += blockDim.x) {
    int r  = idx / WP;
    int cc = idx - r * WP;
    int h = r - 3, w = cc - 3;
    float v = (h >= 0 && h < H_ && w >= 0 && w < W_) ? src[h * W_ + w] : 0.0f;
    dst[idx] = (bf16_t)v;
  }
}

// ---------------------------------------------------------------------------
// Prep 2: per (b,c): 2x2 max-pool 14x14 -> 7x7, softmax over 49, then emit the
// seven 32x16 Toeplitz B-matrices *already in WMMA B-operand VGPR layout*:
//   lane L holds column n = L%16; half e holds K = (e<8 ? e : e+8) + 8*(L>=16);
//   B[K,n] = k[dh, K-n] when 0 <= K-n < 7, else 0.
// Main kernel then fetches B with one contiguous 32B load per lane.
// ---------------------------------------------------------------------------
__global__ void kernel_prep(const float* __restrict__ xk,
                            bf16_t* __restrict__ bws) {
  const int bc = blockIdx.x;
  const float* src = xk + (size_t)bc * (14 * 14);
  __shared__ float sk[49];
  __shared__ float sred;
  const int t = threadIdx.x; // 64 threads
  if (t < 49) {
    int ph = t / 7, pw = t - 7 * (t / 7);
    const float* p = src + (2 * ph) * 14 + 2 * pw;
    sk[t] = fmaxf(fmaxf(p[0], p[1]), fmaxf(p[14], p[15]));
  }
  __syncthreads();
  if (t == 0) {
    float mx = sk[0];
    for (int i = 1; i < 49; ++i) mx = fmaxf(mx, sk[i]);
    sred = mx;
  }
  __syncthreads();
  const float mx = sred;
  if (t < 49) sk[t] = __expf(sk[t] - mx);
  __syncthreads();
  if (t == 0) {
    float s = 0.0f;
    for (int i = 0; i < 49; ++i) s += sk[i];
    sred = 1.0f / s;
  }
  __syncthreads();
  const float inv = sred;

  bf16_t* dst = bws + (size_t)bc * BPKT_HALVES;
  for (int idx = t; idx < BPKT_HALVES; idx += blockDim.x) {
    int dh = idx >> 9;        // /512: which kernel row
    int r  = idx & 511;
    int L  = r >> 4;          // lane
    int e  = r & 15;          // half within lane
    int n  = L & 15;          // output column
    int K  = ((e < 8) ? e : e + 8) + ((L >= 16) ? 8 : 0);
    int d  = K - n;           // tap index dw
    float v = (d >= 0 && d < 7) ? sk[dh * 7 + d] * inv : 0.0f;
    dst[idx] = (bf16_t)v;
  }
}

// ---------------------------------------------------------------------------
// Main fused kernel. One workgroup = one (batch, 16x16 spatial tile) across all
// 256 channels: 16 waves x 16 channels. Per channel: 7 bf16 WMMAs (one per
// kernel row dh) accumulate the conv tile in f32; tiles parked in LDS (256KB),
// Σy / Σy² merged via ds_add_f32; second pass applies LN + ReLU + residual.
// All A/B operand loads are branchless: the upper half-wave's dead K=24..31
// chunk reads the zeroed phantom patch (address select, no EXEC divergence).
// ---------------------------------------------------------------------------
__global__ __launch_bounds__(512, 1)
void dyconv_ln_kernel(const bf16_t* __restrict__ xpad,
                      const bf16_t* __restrict__ bws,
                      const bf16_t* __restrict__ zpatch,
                      const float* __restrict__ x,
                      const float* __restrict__ lnw,
                      const float* __restrict__ lnb,
                      float* __restrict__ out) {
  extern __shared__ char smem[];
  float* ylds = (float*)smem;                        // [C_][16*16] f32 = 256KB
  float* ssum = (float*)(smem + C_ * 256 * 4);       // [256]
  float* ssq  = ssum + 256;                          // [256]

  const int tile = blockIdx.x;                       // 0..35
  const int b    = blockIdx.y;
  const int th = tile / 6, tw = tile - 6 * (tile / 6);
  const int h0 = th * 16, w0 = tw * 16;

  const int tid  = threadIdx.x;
  const int wv   = tid >> 5;
  const int lane = tid & 31;
  const int hi   = lane >> 4;                        // lane group (0/1)
  const int n    = lane & 15;                        // col for B/C/D, row m for A

  // zero the stats accumulators (512 floats, 512 threads; ssum/ssq contiguous)
  ssum[tid] = 0.0f;
  __syncthreads();

  float sreg[8], qreg[8];
#pragma unroll
  for (int j = 0; j < 8; ++j) { sreg[j] = 0.0f; qreg[j] = 0.0f; }

  const int colA  = w0 + (hi ? 8 : 0);               // A chunk-1 column (16B aligned)
  const int colA2 = w0 + 16;                         // A chunk-2 column

  for (int i = 0; i < CH_PER_WAVE; ++i) {
    const int c = wv * CH_PER_WAVE + i;
    const size_t bcp = (size_t)b * C_ + c;
    const bf16_t* xp  = xpad + bcp * PATCH_ELEMS;    // chunk-1 source (real data)
    const bf16_t* xp2 = hi ? zpatch : xp;            // chunk-2: zeros for hi lanes
    const bf16_t* bp  = bws + bcp * BPKT_HALVES + lane * 16;

    if (i + 1 < CH_PER_WAVE) {                       // -> global_prefetch_b8
      __builtin_prefetch(bp + BPKT_HALVES, 0, 1);
      __builtin_prefetch(xp + PATCH_ELEMS + (size_t)(h0 + n) * WP + w0, 0, 1);
    }

    v8f acc0 = {}, acc1 = {};
#pragma unroll
    for (int dh = 0; dh < 7; ++dh) {
      // A operand: lane holds image row m = lane%16; halves map to
      // K = (e<8 ? e : e+8) + 8*hi; only K<22 multiplies nonzero B rows.
      const size_t rowoff = (size_t)(h0 + n + dh) * WP;
      union { v16bf v; v8bf h[2]; } A;
      A.h[0] = *(const v8bf*)(xp  + rowoff + colA);             // K 0..7 / 8..15
      A.h[1] = *(const v8bf*)(xp2 + rowoff + colA2);            // K 16..23 / zeros
      const v16bf Bm = *(const v16bf*)(bp + (size_t)dh * 512);  // pre-packed Toeplitz
      if (dh & 1)
        acc1 = __builtin_amdgcn_wmma_f32_16x16x32_bf16(false, A.v, false, Bm,
                                                       (short)0, acc1, false, false);
      else
        acc0 = __builtin_amdgcn_wmma_f32_16x16x32_bf16(false, A.v, false, Bm,
                                                       (short)0, acc0, false, false);
    }

    // C/D layout: VGPR j, lanes 0-15 -> M=j, lanes 16-31 -> M=j+8; N = lane%16
    float* yc = ylds + c * 256 + hi * 128 + n;
#pragma unroll
    for (int j = 0; j < 8; ++j) {
      float v = acc0[j] + acc1[j];
      yc[j * 16] = v;
      sreg[j] += v;
      qreg[j] += v * v;
    }
  }

  // merge per-wave channel partial sums into per-pixel stats
#pragma unroll
  for (int j = 0; j < 8; ++j) {
    const int p = (j + hi * 8) * 16 + n;
    atomicAdd(&ssum[p], sreg[j]);
    atomicAdd(&ssq[p],  qreg[j]);
  }
  __syncthreads();

  float mu[8], rs[8];
#pragma unroll
  for (int j = 0; j < 8; ++j) {
    const int p = (j + hi * 8) * 16 + n;
    const float s = ssum[p], q = ssq[p];
    const float m = s * (1.0f / C_);
    const float var = q * (1.0f / C_) - m * m;
    mu[j] = m;
    rs[j] = __frsqrt_rn(var + 1e-5f);
  }

  // pass 2: normalize + affine + ReLU + residual, write out (streaming stores)
  for (int i = 0; i < CH_PER_WAVE; ++i) {
    const int c = wv * CH_PER_WAVE + i;
    const float gw = lnw[c];
    const float gb = lnb[c];
    const float* yc = ylds + c * 256 + hi * 128 + n;
    const size_t base = (((size_t)b * C_ + c) * H_ + (h0 + hi * 8)) * W_ + (w0 + n);
    const float* xr = x + base;
    float* orow = out + base;
#pragma unroll
    for (int j = 0; j < 8; ++j) {
      float y  = yc[j * 16];
      float yn = (y - mu[j]) * rs[j] * gw + gb;
      yn = fmaxf(yn, 0.0f);
      float res = __builtin_nontemporal_load(xr + j * W_);
      __builtin_nontemporal_store(yn + res, orow + j * W_);
    }
  }
}

// ---------------------------------------------------------------------------
extern "C" void kernel_launch(void* const* d_in, const int* in_sizes, int n_in,
                              void* d_out, int out_size, void* d_ws, size_t ws_size,
                              hipStream_t stream) {
  const float* x   = (const float*)d_in[0];   // (16,256,96,96) f32
  const float* xk  = (const float*)d_in[1];   // (16,256,14,14) f32
  const float* lnw = (const float*)d_in[2];   // (256,) f32
  const float* lnb = (const float*)d_in[3];   // (256,) f32
  float* out = (float*)d_out;

  // workspace layout: [xpad 86.9MB][bws 29.4MB][zpatch ~21KB]
  bf16_t* xpad = (bf16_t*)d_ws;
  bf16_t* bws  = (bf16_t*)((char*)d_ws +
                           (size_t)(B_ * C_) * PATCH_ELEMS * sizeof(bf16_t));
  bf16_t* zpatch = (bf16_t*)((char*)bws +
                             (size_t)(B_ * C_) * BPKT_HALVES * sizeof(bf16_t));

  pad_convert_kernel<<<B_ * C_ + 1, 256, 0, stream>>>(x, xpad, zpatch);
  kernel_prep<<<B_ * C_, 64, 0, stream>>>(xk, bws);

  dim3 grid(36, B_);                        // 6x6 tiles of 16x16, per batch
  const size_t smem = (size_t)C_ * 256 * 4 + 512 * 4; // 256KB tiles + 2KB stats
  dyconv_ln_kernel<<<grid, 512, smem, stream>>>(xpad, bws, zpatch, x, lnw, lnb, out);
}
